// Encoder_37151467110597
// MI455X (gfx1250) — matrix operand order repaired and checked by reading, not compiled, not run on hardware
//
#include <hip/hip_runtime.h>
#include <math.h>

// ---------------- constants ----------------
#define L_   3
#define H_   4
#define D_   510
#define Dp_  512
#define B_   256
#define N_   100
#define Np_  112          // padded seq for q/k & mask
#define NV_  128          // padded seq for v^T rows (K of attn.V gemm)
#define BN_  (B_*N_)      // 25600
#define HD_  (H_*D_)      // 2040
#define HDp_ (H_*Dp_)     // 2048

typedef __bf16 bf16x16 __attribute__((ext_vector_type(16)));
typedef __bf16 bf16x8  __attribute__((ext_vector_type(8)));
typedef float  f32x8   __attribute__((ext_vector_type(8)));

#define WMMA_BF16(a,b,c) __builtin_amdgcn_wmma_f32_16x16x32_bf16(false,(a),false,(b),(short)0,(c),false,false)

__device__ inline f32x8 zero8() {
  f32x8 z;
  #pragma unroll
  for (int i=0;i<8;i++) z[i]=0.f;
  return z;
}

// A fragment: lane holds row (lane&15), 16 contiguous K elements (k = 16*half + s)
__device__ inline bf16x16 load_a_frag(const __bf16* p) {
  bf16x8 lo = *(const bf16x8*)p;
  bf16x8 hi = *(const bf16x8*)(p + 8);
  bf16x16 r;
  #pragma unroll
  for (int i=0;i<8;i++){ r[i]=lo[i]; r[8+i]=hi[i]; }
  return r;
}

// ---------------- helper reductions ----------------
__device__ inline float block_sum_128(float v, float* sm) {
  #pragma unroll
  for (int m=16;m>=1;m>>=1) v += __shfl_xor(v, m, 32);
  int w = threadIdx.x >> 5;
  if ((threadIdx.x & 31) == 0) sm[w] = v;
  __syncthreads();
  v = sm[0] + sm[1] + sm[2] + sm[3];
  __syncthreads();
  return v;
}

// ---------------- small utility kernels ----------------
__global__ void zero_f32_kernel(float* p, int n) {
  int i = blockIdx.x * blockDim.x + threadIdx.x;
  if (i < n) p[i] = 0.f;
}

// h = x + PE, mask[b,n] = (sum_d x != 0), outputs f32 + bf16 (cols 510,511 zero)
__global__ __launch_bounds__(128)
void embed_kernel(const float* __restrict__ x, float* __restrict__ h32,
                  __bf16* __restrict__ hbf, float* __restrict__ maskf) {
  __shared__ float sm[4];
  const int r = blockIdx.x, tid = threadIdx.x;
  const int b = r / N_, n = r % N_;
  const float* xr = x + (size_t)r * D_;
  float s = 0.f;
  for (int d = tid; d < D_; d += 128) s += xr[d];
  s = block_sum_128(s, sm);
  if (tid == 0) maskf[(size_t)b * Np_ + n] = (s != 0.f) ? 1.f : 0.f;
  const float c = -9.210340371976184f / (float)D_;   // -ln(10000)/D
  #pragma unroll
  for (int i = 0; i < 4; i++) {
    int d = tid + i * 128;
    float v = 0.f;
    if (d < D_) {
      int i2 = d & ~1;
      float ang = (float)n * __expf((float)i2 * c);
      float pe = (d & 1) ? __cosf(ang) : __sinf(ang);
      v = xr[d] + pe;
    }
    h32[(size_t)r * Dp_ + d] = v;
    hbf[(size_t)r * Dp_ + d] = (__bf16)v;
  }
}

// dst[c*Rp + kk] = bf16(src[r*C + c]); kk=(r/seg_len)*seg_pad + r%seg_len (zero-pad)
__global__ void transpose_bf16_kernel(const float* __restrict__ src, __bf16* __restrict__ dst,
                                      int R, int C, int Rp, int Cp, int seg_len, int seg_pad) {
  int idx = blockIdx.x * blockDim.x + threadIdx.x;
  if (idx >= Cp * Rp) return;
  int c = idx / Rp, kpos = idx % Rp;
  int seg = kpos / seg_pad, off = kpos % seg_pad;
  int rr = seg * seg_len + off;
  float v = (off < seg_len && rr < R && c < C) ? src[(size_t)rr * C + c] : 0.f;
  dst[idx] = (__bf16)v;
}

// ---------------- generic WMMA bf16 GEMM ----------------
// C[M x 64-per-block] = A[M x Kp](bf16,row-major) * BT[N x Kp](bf16,row-major as B^T) + bias
// out_kind 0: bf16 with batch remap  out + b*bstride + rr*ldo + n
// out_kind 1: bf16 transposed        out + b*bstride + n*ldo  + rr
// out_kind 2: f32 plain              out + m*ldo + n
__global__ __launch_bounds__(32)
void gemm_bf16_kernel(const __bf16* __restrict__ Amat, int lda,
                      const __bf16* __restrict__ BT, int ldb,
                      const float* __restrict__ bias, int Nreal, int Kp,
                      int out_kind, void* __restrict__ Out, int ldo,
                      int rows_per_b, long long out_bstride) {
  const int lane = threadIdx.x & 31;
  const int lr = lane & 15, hs = lane >> 4;
  const int mt = blockIdx.x, nb = blockIdx.y;
  f32x8 acc[4];
  #pragma unroll
  for (int g=0; g<4; g++) acc[g] = zero8();

  const __bf16* arow = Amat + (size_t)(mt*16 + lr) * lda + hs * 16;
  const int n0 = nb * 64;
  const __bf16* brow = BT + (size_t)(n0 + lr) * ldb + hs * 16;

  for (int k = 0; k < Kp; k += 32) {
    bf16x16 af = load_a_frag(arow + k);
    #pragma unroll
    for (int g = 0; g < 4; g++) {
      bf16x16 bf_ = *(const bf16x16*)(brow + (size_t)g * 16 * ldb + k);
      acc[g] = WMMA_BF16(af, bf_, acc[g]);
    }
  }

  #pragma unroll
  for (int g = 0; g < 4; g++) {
    int n = n0 + g * 16 + lr;
    float bv = (bias != nullptr && n < Nreal) ? bias[n] : 0.f;
    #pragma unroll
    for (int j = 0; j < 8; j++) {
      int m = mt * 16 + hs * 8 + j;
      float val = (n < Nreal) ? (acc[g][j] + bv) : 0.f;
      int bb = m / rows_per_b, rr = m % rows_per_b;
      if (out_kind == 0) {
        ((__bf16*)Out)[(size_t)bb * out_bstride + (size_t)rr * ldo + n] = (__bf16)val;
      } else if (out_kind == 1) {
        ((__bf16*)Out)[(size_t)bb * out_bstride + (size_t)n * ldo + rr] = (__bf16)val;
      } else {
        ((float*)Out)[(size_t)m * ldo + n] = val;
      }
    }
  }
}

// ---------------- fused attention (dots -> mask -> softmax -> attn.V) ----------------
// grid: (7 n-tiles, B*H); one wave per block.
__global__ __launch_bounds__(32)
void attention_kernel(const __bf16* __restrict__ q, const __bf16* __restrict__ k,
                      const __bf16* __restrict__ vT, const float* __restrict__ maskf,
                      __bf16* __restrict__ obf, float scale) {
  __shared__ __bf16 att[16 * NV_];
  const int lane = threadIdx.x & 31;
  const int lr = lane & 15, hs = lane >> 4;
  const int nt = blockIdx.x;
  const int bh = blockIdx.y;
  const int b = bh / H_, head = bh % H_;

  const __bf16* qb = q  + (size_t)bh * Np_ * Dp_;
  const __bf16* kb = k  + (size_t)bh * Np_ * Dp_;
  const __bf16* vb = vT + (size_t)bh * Dp_ * NV_;

  // ---- dots = q . k^T  (A rows = q rows, B cols = k rows) ----
  f32x8 acc[7];
  #pragma unroll
  for (int t=0;t<7;t++) acc[t] = zero8();
  const __bf16* arow = qb + (size_t)(nt*16 + lr) * Dp_ + hs * 16;
  for (int kk = 0; kk < Dp_; kk += 32) {
    bf16x16 af = load_a_frag(arow + kk);
    #pragma unroll
    for (int t = 0; t < 7; t++) {
      bf16x16 bf_ = *(const bf16x16*)(kb + (size_t)(t*16 + lr) * Dp_ + hs*16 + kk);
      acc[t] = WMMA_BF16(af, bf_, acc[t]);
    }
  }

  // ---- mask + scale ----
  float mrow[8], mcol[7];
  #pragma unroll
  for (int j=0;j<8;j++) mrow[j] = maskf[(size_t)b * Np_ + nt*16 + hs*8 + j];
  #pragma unroll
  for (int t=0;t<7;t++) mcol[t] = maskf[(size_t)b * Np_ + t*16 + lr];
  #pragma unroll
  for (int t=0;t<7;t++) {
    bool real = (t*16 + lr) < N_;
    #pragma unroll
    for (int j=0;j<8;j++) {
      float v = acc[t][j];
      v = (mcol[t] != 0.f && mrow[j] != 0.f) ? v * scale
                                             : (real ? -3.0e38f : -INFINITY);
      acc[t][j] = v;
    }
  }

  // ---- softmax over columns (lanes within each 16-lane half + 7 tiles) ----
  #pragma unroll
  for (int j=0;j<8;j++) {
    float mx = -INFINITY;
    #pragma unroll
    for (int t=0;t<7;t++) mx = fmaxf(mx, acc[t][j]);
    #pragma unroll
    for (int m=8;m>=1;m>>=1) mx = fmaxf(mx, __shfl_xor(mx, m, 32));
    float e[7]; float s = 0.f;
    #pragma unroll
    for (int t=0;t<7;t++) { e[t] = __expf(acc[t][j] - mx); s += e[t]; }
    #pragma unroll
    for (int m=8;m>=1;m>>=1) s += __shfl_xor(s, m, 32);
    float inv = 1.f / s;
    #pragma unroll
    for (int t=0;t<7;t++)
      att[(hs*8 + j) * NV_ + t*16 + lr] = (__bf16)(e[t] * inv);
  }
  // zero pad columns 112..127
  for (int idx = lane; idx < 256; idx += 32)
    att[(idx >> 4) * NV_ + Np_ + (idx & 15)] = (__bf16)0.f;
  __syncthreads();

  // ---- o = attn . v  (A from LDS, B rows = vT rows, contiguous) ----
  for (int dt = 0; dt < Dp_/16; dt++) {
    f32x8 oc = zero8();
    #pragma unroll
    for (int ks = 0; ks < NV_/32; ks++) {
      bf16x16 af = load_a_frag(&att[lr * NV_ + ks*32 + hs*16]);
      bf16x16 bf_ = *(const bf16x16*)(vb + (size_t)(dt*16 + lr) * NV_ + ks*32 + hs*16);
      oc = WMMA_BF16(af, bf_, oc);
    }
    int d = dt*16 + lr;                 // output column (lane)
    #pragma unroll
    for (int j=0;j<8;j++) {
      int n = nt*16 + hs*8 + j;
      if (n < N_) {
        size_t r = (size_t)b * N_ + n;
        obf[r * HDp_ + head * Dp_ + d] = (__bf16)oc[j];
      }
    }
  }
}

// ---------------- bias/residual/(gelu)/LayerNorm ----------------
__global__ __launch_bounds__(128)
void ln_kernel(const float* __restrict__ in, const float* __restrict__ resid,
               const float* __restrict__ gamma, const float* __restrict__ beta,
               float* __restrict__ outF, __bf16* __restrict__ outB, int do_gelu) {
  __shared__ float sm[4];
  const int r = blockIdx.x, tid = threadIdx.x;
  float xa[4]; float s = 0.f, sq = 0.f;
  #pragma unroll
  for (int i=0;i<4;i++) {
    int d = tid + i*128;
    float x = 0.f;
    if (d < D_) {
      x = in[(size_t)r * Dp_ + d];
      if (do_gelu) x = 0.5f * x * (1.f + erff(x * 0.7071067811865475f));
      x += resid[(size_t)r * Dp_ + d];
    }
    xa[i] = x; s += x; sq += x * x;
  }
  s  = block_sum_128(s,  sm);
  sq = block_sum_128(sq, sm);
  float mean = s / (float)D_;
  float var  = sq / (float)D_ - mean * mean;
  float invd = rsqrtf(var + 1e-5f);
  #pragma unroll
  for (int i=0;i<4;i++) {
    int d = tid + i*128;
    float y = (d < D_) ? (xa[i] - mean) * invd * gamma[d] + beta[d] : 0.f;
    outF[(size_t)r * Dp_ + d] = y;
    outB[(size_t)r * Dp_ + d] = (__bf16)y;
  }
}

__global__ void compact_kernel(const float* __restrict__ h32, float* __restrict__ out, int total) {
  int i = blockIdx.x * blockDim.x + threadIdx.x;
  if (i >= total) return;
  int r = i / D_, d = i % D_;
  out[i] = h32[(size_t)r * Dp_ + d];
}

// ---------------- host orchestration ----------------
extern "C" void kernel_launch(void* const* d_in, const int* in_sizes, int n_in,
                              void* d_out, int out_size, void* d_ws, size_t ws_size,
                              hipStream_t stream) {
  const float* x   = (const float*)d_in[0];
  const float* Wq  = (const float*)d_in[1];
  const float* bq  = (const float*)d_in[2];
  const float* Wk  = (const float*)d_in[3];
  const float* bk  = (const float*)d_in[4];
  const float* Wv  = (const float*)d_in[5];
  const float* bv  = (const float*)d_in[6];
  const float* Wo  = (const float*)d_in[7];
  const float* bo  = (const float*)d_in[8];
  const float* Wf  = (const float*)d_in[9];
  const float* bfb = (const float*)d_in[10];
  const float* g1  = (const float*)d_in[11];
  const float* b1  = (const float*)d_in[12];
  const float* g2  = (const float*)d_in[13];
  const float* b2  = (const float*)d_in[14];
  float* out = (float*)d_out;

  char* wsb = (char*)d_ws; size_t off = 0;
  auto alloc = [&](size_t bytes) -> char* {
    char* p = wsb + off; off = (off + bytes + 255) & ~(size_t)255; return p;
  };
  __bf16* hbf   = (__bf16*)alloc((size_t)BN_ * Dp_ * 2);
  float*  h32   = (float*) alloc((size_t)BN_ * Dp_ * 4);
  float*  maskf = (float*) alloc((size_t)B_ * Np_ * 4);
  __bf16* WqT   = (__bf16*)alloc((size_t)L_*H_ * Dp_ * Dp_ * 2);
  __bf16* WkT   = (__bf16*)alloc((size_t)L_*H_ * Dp_ * Dp_ * 2);
  __bf16* WvT   = (__bf16*)alloc((size_t)L_*H_ * Dp_ * Dp_ * 2);
  __bf16* WoT   = (__bf16*)alloc((size_t)L_ * Dp_ * HDp_ * 2);
  __bf16* WfT   = (__bf16*)alloc((size_t)L_ * Dp_ * Dp_ * 2);
  __bf16* qbuf  = (__bf16*)alloc((size_t)B_ * H_ * Np_ * Dp_ * 2);
  __bf16* kbuf  = (__bf16*)alloc((size_t)B_ * H_ * Np_ * Dp_ * 2);
  __bf16* vtb   = (__bf16*)alloc((size_t)B_ * H_ * Dp_ * NV_ * 2);
  __bf16* obf   = (__bf16*)alloc((size_t)BN_ * HDp_ * 2);
  float*  abuf  = (float*) alloc((size_t)BN_ * Dp_ * 4);
  float*  ln1f  = (float*) alloc((size_t)BN_ * Dp_ * 4);
  __bf16* ln1b  = (__bf16*)alloc((size_t)BN_ * Dp_ * 2);

  const float scale = 1.0f / sqrtf((float)D_);

  // mask padding + embedding
  zero_f32_kernel<<<(B_*Np_ + 255)/256, 256, 0, stream>>>(maskf, B_*Np_);
  embed_kernel<<<BN_, 128, 0, stream>>>(x, h32, hbf, maskf);

  // weight transpose + bf16 conversion
  {
    int tgrid = (Dp_*Dp_ + 255)/256;
    for (int s = 0; s < L_*H_; s++) {
      transpose_bf16_kernel<<<tgrid, 256, 0, stream>>>(Wq + (size_t)s*D_*D_, WqT + (size_t)s*Dp_*Dp_, D_, D_, Dp_, Dp_, D_, Dp_);
      transpose_bf16_kernel<<<tgrid, 256, 0, stream>>>(Wk + (size_t)s*D_*D_, WkT + (size_t)s*Dp_*Dp_, D_, D_, Dp_, Dp_, D_, Dp_);
      transpose_bf16_kernel<<<tgrid, 256, 0, stream>>>(Wv + (size_t)s*D_*D_, WvT + (size_t)s*Dp_*Dp_, D_, D_, Dp_, Dp_, D_, Dp_);
    }
    int ogrid = (Dp_*HDp_ + 255)/256;
    for (int l = 0; l < L_; l++) {
      transpose_bf16_kernel<<<ogrid, 256, 0, stream>>>(Wo + (size_t)l*HD_*D_, WoT + (size_t)l*Dp_*HDp_, HD_, D_, HDp_, Dp_, D_, Dp_);
      transpose_bf16_kernel<<<tgrid, 256, 0, stream>>>(Wf + (size_t)l*D_*D_,  WfT + (size_t)l*Dp_*Dp_,  D_, D_, Dp_, Dp_, D_, Dp_);
    }
  }

  dim3 ggrid(BN_/16, Dp_/64);          // 1600 x 8
  for (int l = 0; l < L_; l++) {
    // per-head projections
    for (int h = 0; h < H_; h++) {
      int s = l*H_ + h;
      gemm_bf16_kernel<<<ggrid, 32, 0, stream>>>(hbf, Dp_, WqT + (size_t)s*Dp_*Dp_, Dp_,
          bq + (size_t)s*D_, D_, Dp_, 0, qbuf + (size_t)h*Np_*Dp_, Dp_, N_, (long long)H_*Np_*Dp_);
      gemm_bf16_kernel<<<ggrid, 32, 0, stream>>>(hbf, Dp_, WkT + (size_t)s*Dp_*Dp_, Dp_,
          bk + (size_t)s*D_, D_, Dp_, 0, kbuf + (size_t)h*Np_*Dp_, Dp_, N_, (long long)H_*Np_*Dp_);
      gemm_bf16_kernel<<<ggrid, 32, 0, stream>>>(hbf, Dp_, WvT + (size_t)s*Dp_*Dp_, Dp_,
          bv + (size_t)s*D_, D_, Dp_, 1, vtb + (size_t)h*Dp_*NV_, NV_, N_, (long long)H_*Dp_*NV_);
    }
    // attention
    attention_kernel<<<dim3(Np_/16, B_*H_), 32, 0, stream>>>(qbuf, kbuf, vtb, maskf, obf, scale);
    // output projection (f32 out)
    gemm_bf16_kernel<<<ggrid, 32, 0, stream>>>(obf, HDp_, WoT + (size_t)l*Dp_*HDp_, HDp_,
        bo + (size_t)l*D_, D_, HDp_, 2, abuf, Dp_, BN_, 0);
    // ln1 = LN(a + h)
    ln_kernel<<<BN_, 128, 0, stream>>>(abuf, h32, g1 + (size_t)l*D_, b1 + (size_t)l*D_, ln1f, ln1b, 0);
    // FFN (f32 out)
    gemm_bf16_kernel<<<ggrid, 32, 0, stream>>>(ln1b, Dp_, WfT + (size_t)l*Dp_*Dp_, Dp_,
        bfb + (size_t)l*D_, D_, Dp_, 2, abuf, Dp_, BN_, 0);
    // h = LN(gelu(f) + ln1)
    ln_kernel<<<BN_, 128, 0, stream>>>(abuf, ln1f, g2 + (size_t)l*D_, b2 + (size_t)l*D_, h32, hbf, 1);
  }

  compact_kernel<<<(BN_*D_ + 255)/256, 256, 0, stream>>>(h32, out, BN_*D_);
}
